// LlamaMaverickAttention_7164005449794
// MI455X (gfx1250) — matrix operand order repaired and checked by reading, not compiled, not run on hardware
//
#include <hip/hip_runtime.h>
#include <stdint.h>

typedef __bf16 bf16_t;
typedef __attribute__((ext_vector_type(16))) __bf16 v16bf;
typedef __attribute__((ext_vector_type(8)))  float  v8f;
typedef __attribute__((ext_vector_type(4)))  unsigned int v4u;
typedef __attribute__((ext_vector_type(8)))  int  v8i;
typedef __attribute__((ext_vector_type(4)))  int  v4i;

#if defined(__has_builtin)
#if __has_builtin(__builtin_amdgcn_tensor_load_to_lds)
#define HAVE_TDM 1
#endif
#endif

union FragB16 { v16bf v; uint32_t u[8]; };

__device__ __forceinline__ bf16_t f2bf(float f) {
  uint32_t u = __builtin_bit_cast(uint32_t, f);
  uint32_t r = u + 0x7FFFu + ((u >> 16) & 1u);
  unsigned short h = (unsigned short)(r >> 16);
  return __builtin_bit_cast(bf16_t, h);
}
__device__ __forceinline__ float bf2f(bf16_t b) {
  unsigned short h = __builtin_bit_cast(unsigned short, b);
  uint32_t u = ((uint32_t)h) << 16;
  return __builtin_bit_cast(float, u);
}
__device__ __forceinline__ uint32_t pack2(float a, float b) {
  return (uint32_t)__builtin_bit_cast(unsigned short, f2bf(a)) |
         ((uint32_t)__builtin_bit_cast(unsigned short, f2bf(b)) << 16);
}

// DPP xor-shuffle within 16-lane rows (VALU, keeps the DS pipe free for frag loads)
template <int CTRL>
__device__ __forceinline__ float dpp_xf(float x) {
  int v = __builtin_bit_cast(int, x);
  int r = __builtin_amdgcn_update_dpp(v, v, CTRL, 0xf, 0xf, false);
  return __builtin_bit_cast(float, r);
}
__device__ __forceinline__ float rowmax16(float v) {
  v = fmaxf(v, dpp_xf<0xB1>(v));    // quad_perm(1,0,3,2) : xor 1
  v = fmaxf(v, dpp_xf<0x4E>(v));    // quad_perm(2,3,0,1) : xor 2
  v = fmaxf(v, dpp_xf<0x164>(v));   // ROW_XMASK 4
  v = fmaxf(v, dpp_xf<0x168>(v));   // ROW_XMASK 8
  return v;
}
__device__ __forceinline__ float rowsum16(float v) {
  v += dpp_xf<0xB1>(v);
  v += dpp_xf<0x4E>(v);
  v += dpp_xf<0x164>(v);
  v += dpp_xf<0x168>(v);
  return v;
}

// 16-byte global -> LDS async copy (ASYNCcnt-tracked, no VGPR landing).
__device__ __forceinline__ void async_copy16(void* lds_dst, const void* gsrc) {
  uint32_t l = (uint32_t)(uintptr_t)lds_dst;
  uint64_t g = (uint64_t)(uintptr_t)gsrc;
  asm volatile("global_load_async_to_lds_b128 %0, %1, off" :: "v"(l), "v"(g) : "memory");
}
__device__ __forceinline__ void wait_async0() {
  asm volatile("s_wait_asynccnt 0x0" ::: "memory");
}

#ifdef HAVE_TDM
// Tensor Data Mover: DMA a [tile_h x tile_w] bf16 tile (row stride in elems) from
// global into LDS. Wave-uniform, TENSORcnt-tracked. D# per CDNA5 ISA 8.3/8.4.
__device__ __forceinline__ void tdm_load_2d(void* lds_dst, const void* gsrc,
                                            int tile_w, int tile_h, int row_stride) {
  uint64_t ga = (uint64_t)(uintptr_t)gsrc;
  v4u g0;
  g0[0] = 1u;                                   // count=1, user mode
  g0[1] = (uint32_t)(uintptr_t)lds_dst;         // lds_addr (bytes)
  g0[2] = (uint32_t)ga;                         // global_addr[31:0]
  g0[3] = (uint32_t)(ga >> 32) | (2u << 30);    // global_addr[56:32] | type=2
  v8i g1;
  g1[0] = (int)0x00010000u;                     // wg_mask=0 | data_size=1 (2 bytes)
  g1[1] = (int)((uint32_t)row_stride << 16);    // tensor_dim0[15:0]
  g1[2] = (int)(((uint32_t)row_stride >> 16) | (0xFFFFu << 16)); // dim0 hi | dim1 lo
  g1[3] = (int)((uint32_t)tile_w << 16);        // tensor_dim1 hi=0 | tile_dim0
  g1[4] = tile_h;                               // tile_dim1 | tile_dim2=0
  g1[5] = row_stride;                           // tensor_dim0_stride[31:0]
  g1[6] = 0;
  g1[7] = 0;
  v4i z4; z4[0] = 0; z4[1] = 0; z4[2] = 0; z4[3] = 0;
  v8i z8;
#pragma unroll
  for (int i = 0; i < 8; ++i) z8[i] = 0;
  __builtin_amdgcn_tensor_load_to_lds(g0, g1, z4, z4, z8, 0);
}
#endif

// ---------------- fp32 -> bf16 convert (4 elems / thread) ----------------
__global__ void k_f32_to_bf16(const float* __restrict__ src, bf16_t* __restrict__ dst, int n4) {
  int i = blockIdx.x * blockDim.x + threadIdx.x;
  if (i >= n4) return;
  float4 f = ((const float4*)src)[i];
  uint2 r; r.x = pack2(f.x, f.y); r.y = pack2(f.z, f.w);
  ((uint2*)dst)[i] = r;
}

// ---------- concat-convert [wq;wk;wv] (fp32) -> bf16 [3072,2048] ----------
__global__ void k_pack_qkv_w(const float* __restrict__ wq, const float* __restrict__ wk,
                             const float* __restrict__ wv, bf16_t* __restrict__ dst,
                             int nq, int nk, int nv) {
  int i = blockIdx.x * blockDim.x + threadIdx.x;
  long long e = (long long)i * 4;
  long long total = (long long)nq + nk + nv;
  if (e >= total) return;
  const float* src; long long off;
  if (e < nq)            { src = wq; off = e; }
  else if (e < nq + nk)  { src = wk; off = e - nq; }
  else                   { src = wv; off = e - nq - nk; }
  float4 f = *(const float4*)(src + off);
  uint2 r; r.x = pack2(f.x, f.y); r.y = pack2(f.z, f.w);
  ((uint2*)dst)[i] = r;
}

// ---------------- bf16 WMMA GEMM: C[M,N] = A[M,K] * W[N,K]^T ----------------
// 256 threads (8 waves), block tile 128x128, wave tile 32x64, k-step 32,
// double-buffered LDS with TDM (or async-copy) prefetch of the next k-tile.
template <bool OUT_F32>
__global__ __launch_bounds__(256) void k_gemm_bf16(const bf16_t* __restrict__ A,
                                                   const bf16_t* __restrict__ W,
                                                   void* __restrict__ Cout,
                                                   int M, int N, int K) {
  __shared__ bf16_t ldsA[2][128 * 32];
  __shared__ bf16_t ldsW[2][128 * 32];
  const int tid  = threadIdx.x;
  const int lane = tid & 31;
  const int wid  = tid >> 5;
  const int wm   = wid & 3;
  const int wn   = wid >> 2;
  const int bm   = blockIdx.x * 128;
  const int bn   = blockIdx.y * 128;
  const int nl   = lane & 15;
  const int half = lane >> 4;

  auto stage = [&](int k0, int buf) {
#ifdef HAVE_TDM
    if (wid == 0) {
      tdm_load_2d(ldsA[buf], &A[(size_t)bm * K + k0], 32, 128, K);
      tdm_load_2d(ldsW[buf], &W[(size_t)bn * K + k0], 32, 128, K);
    }
#else
#pragma unroll
    for (int c = tid; c < 512; c += 256) {
      int row = c >> 2;
      int col = (c & 3) * 8;
      async_copy16(&ldsA[buf][row * 32 + col], &A[(size_t)(bm + row) * K + k0 + col]);
      async_copy16(&ldsW[buf][row * 32 + col], &W[(size_t)(bn + row) * K + k0 + col]);
    }
#endif
  };

  v8f acc[2][4];
#pragma unroll
  for (int mi = 0; mi < 2; ++mi)
#pragma unroll
    for (int ni = 0; ni < 4; ++ni)
#pragma unroll
      for (int r = 0; r < 8; ++r) acc[mi][ni][r] = 0.f;

  stage(0, 0);
  for (int k0 = 0; k0 < K; k0 += 32) {
    const int cur = (k0 >> 5) & 1;
#ifdef HAVE_TDM
    if (wid == 0) __builtin_amdgcn_s_wait_tensorcnt(0);
#else
    wait_async0();
#endif
    __syncthreads();
    if (k0 + 32 < K) stage(k0 + 32, cur ^ 1);   // prefetch next tile during compute

    FragB16 af[2], bf[4];
#pragma unroll
    for (int mi = 0; mi < 2; ++mi) {
      int m = wm * 32 + mi * 16 + nl;
#pragma unroll
      for (int j = 0; j < 8; ++j) {
        int k = half * 16 + 2 * j;                        // A-frag K pattern
        af[mi].u[j] = *(const uint32_t*)&ldsA[cur][m * 32 + k];
      }
    }
#pragma unroll
    for (int ni = 0; ni < 4; ++ni) {
      int n = wn * 64 + ni * 16 + nl;
#pragma unroll
      for (int j = 0; j < 8; ++j) {
        int k = half * 16 + 2 * j;                        // B-frag K pattern
        bf[ni].u[j] = *(const uint32_t*)&ldsW[cur][n * 32 + k];
      }
    }
#pragma unroll
    for (int ni = 0; ni < 4; ++ni)
#pragma unroll
      for (int mi = 0; mi < 2; ++mi)
        acc[mi][ni] = __builtin_amdgcn_wmma_f32_16x16x32_bf16(
            false, af[mi].v, false, bf[ni].v, (short)0, acc[mi][ni], false, false);
  }

#pragma unroll
  for (int mi = 0; mi < 2; ++mi)
#pragma unroll
    for (int ni = 0; ni < 4; ++ni)
#pragma unroll
      for (int r = 0; r < 8; ++r) {
        int row = bm + wm * 32 + mi * 16 + half * 8 + r;
        int col = bn + wn * 64 + ni * 16 + nl;
        float v = acc[mi][ni][r];
        if (OUT_F32) ((float*)Cout)[(size_t)row * N + col] = v;
        else         ((bf16_t*)Cout)[(size_t)row * N + col] = f2bf(v);
      }
}

// -------- RoPE in place on Q/K sections of QKV [S,3072]; folds 1/sqrt(d) into Q --------
__global__ void k_rope(bf16_t* __restrict__ qkv, int S) {
  int t = blockIdx.x * blockDim.x + threadIdx.x;
  if (t >= S * 20 * 64) return;
  int j    = t & 63;
  int t2   = t >> 6;
  int head = t2 % 20;                  // 0..15 q heads, 16..19 k heads
  int s    = t2 / 20;
  bf16_t* p = qkv + (size_t)s * 3072 +
              (head < 16 ? head * 128 : 2048 + (head - 16) * 128);
  float os = (head < 16) ? 0.08838834764831845f : 1.0f;  // 1/sqrt(128) folded into Q
  float freq = __expf(-(float)(2 * j) * (9.210340371976184f / 128.f)); // 10000^(-2j/128)
  float th = (float)s * freq;
  float c = __cosf(th), sn = __sinf(th);
  float x1 = bf2f(p[j]), x2 = bf2f(p[j + 64]);
  p[j]      = f2bf((x1 * c - x2 * sn) * os);
  p[j + 64] = f2bf((x2 * c + x1 * sn) * os);
}

// ---------------- flash attention: block = (head, 64 queries), 4 waves ----------------
// Double-buffered K/V tiles; K via TDM DMA, V transposed during manual staging.
__global__ __launch_bounds__(128) void k_attn(const bf16_t* __restrict__ qkv,
                                              bf16_t* __restrict__ out, int S) {
  __shared__ bf16_t ldsK[2][32 * 128];   // [key][d]
  __shared__ bf16_t ldsVt[2][128 * 32];  // [d][key]
  __shared__ bf16_t ldsP[4 * 16 * 32];   // per-wave P scratch

  const int tid  = threadIdx.x;
  const int lane = tid & 31;
  const int w    = tid >> 5;
  const int nl   = lane & 15;
  const int half = lane >> 4;

  const int qtiles = S / 64;
  const int h   = blockIdx.x / qtiles;
  const int qt  = blockIdx.x % qtiles;
  const int kvh = h >> 2;

  const bf16_t* Qb = qkv + h * 128;            // row stride 3072
  const bf16_t* Kb = qkv + 2048 + kvh * 128;
  const bf16_t* Vb = qkv + 2560 + kvh * 128;
  const int m0 = qt * 64 + w * 16;

  auto stageKV = [&](int kb, int buf) {
#ifdef HAVE_TDM
    if (w == 0) tdm_load_2d(ldsK[buf], &Kb[(size_t)kb * 3072], 128, 32, 3072);
#endif
#pragma unroll
    for (int c = tid; c < 512; c += 128) {
      int key = c >> 4;
      int d0  = (c & 15) * 8;
#ifndef HAVE_TDM
      async_copy16(&ldsK[buf][key * 128 + d0], &Kb[(size_t)(kb + key) * 3072 + d0]);
#endif
      uint4 vv = *(const uint4*)&Vb[(size_t)(kb + key) * 3072 + d0];
      const bf16_t* ve = (const bf16_t*)&vv;
#pragma unroll
      for (int i = 0; i < 8; ++i) ldsVt[buf][(d0 + i) * 32 + key] = ve[i];
    }
  };

  // preload Q fragments for all 4 d-chunks of 32 (already scaled by 1/sqrt(d))
  FragB16 qf[4];
#pragma unroll
  for (int c = 0; c < 4; ++c) {
    const bf16_t* row = Qb + (size_t)(m0 + nl) * 3072 + c * 32;
#pragma unroll
    for (int j = 0; j < 8; ++j) {
      int k = half * 16 + 2 * j;
      qf[c].u[j] = *(const uint32_t*)&row[k];
    }
  }

  float row_m[8], row_l[8];
  v8f o[8];
#pragma unroll
  for (int r = 0; r < 8; ++r) { row_m[r] = -3.0e38f; row_l[r] = 0.f; }
#pragma unroll
  for (int t8 = 0; t8 < 8; ++t8)
#pragma unroll
    for (int r = 0; r < 8; ++r) o[t8][r] = 0.f;

  stageKV(0, 0);
  for (int kb = 0; kb < S; kb += 32) {
    const int cur = (kb >> 5) & 1;
#ifdef HAVE_TDM
    if (w == 0) __builtin_amdgcn_s_wait_tensorcnt(0);
#else
    wait_async0();
#endif
    __syncthreads();
    if (kb + 32 < S) stageKV(kb + 32, cur ^ 1);   // prefetch next K/V tile

    // ---- load ALL K B-frags first, then burst the WMMAs ----
    FragB16 bk0[4], bk1[4];
#pragma unroll
    for (int c = 0; c < 4; ++c) {
#pragma unroll
      for (int j = 0; j < 8; ++j) {
        bk0[c].u[j] = *(const uint32_t*)&ldsK[cur][(nl)      * 128 + c * 32 + half * 16 + 2 * j];
        bk1[c].u[j] = *(const uint32_t*)&ldsK[cur][(16 + nl) * 128 + c * 32 + half * 16 + 2 * j];
      }
    }
    v8f sa0, sa1;
#pragma unroll
    for (int r = 0; r < 8; ++r) { sa0[r] = 0.f; sa1[r] = 0.f; }
#pragma unroll
    for (int c = 0; c < 4; ++c) {
      sa0 = __builtin_amdgcn_wmma_f32_16x16x32_bf16(false, qf[c].v, false, bk0[c].v,
                                                    (short)0, sa0, false, false);
      sa1 = __builtin_amdgcn_wmma_f32_16x16x32_bf16(false, qf[c].v, false, bk1[c].v,
                                                    (short)0, sa1, false, false);
    }

    // online softmax per row (rows live in 16-lane halves; DPP xor-reductions)
    float p0[8], p1[8], scl[8];
#pragma unroll
    for (int r = 0; r < 8; ++r) {
      float s0 = sa0[r], s1 = sa1[r];
      float mx = rowmax16(fmaxf(s0, s1));
      float mnew = fmaxf(row_m[r], mx);
      float sc = __expf(row_m[r] - mnew);
      row_m[r] = mnew;
      float e0 = __expf(s0 - mnew), e1 = __expf(s1 - mnew);
      float rs = rowsum16(e0 + e1);
      row_l[r] = row_l[r] * sc + rs;
      p0[r] = e0; p1[r] = e1; scl[r] = sc;
    }
#pragma unroll
    for (int t8 = 0; t8 < 8; ++t8)
#pragma unroll
      for (int r = 0; r < 8; ++r) o[t8][r] *= scl[r];

    // P (C-layout) -> per-wave LDS -> A-fragment
    bf16_t* Pw = ldsP + w * 512;
#pragma unroll
    for (int r = 0; r < 8; ++r) {
      int m = half * 8 + r;
      Pw[m * 32 + nl]      = f2bf(p0[r]);
      Pw[m * 32 + 16 + nl] = f2bf(p1[r]);
    }
    FragB16 pf;
#pragma unroll
    for (int j = 0; j < 8; ++j) {
      int k = half * 16 + 2 * j;
      pf.u[j] = *(const uint32_t*)&Pw[nl * 32 + k];
    }

    // O += P * V : preload V frags in groups of 4, then WMMA bursts
#pragma unroll
    for (int g = 0; g < 2; ++g) {
      FragB16 bv[4];
#pragma unroll
      for (int t4 = 0; t4 < 4; ++t4)
#pragma unroll
        for (int j = 0; j < 8; ++j)
          bv[t4].u[j] =
              *(const uint32_t*)&ldsVt[cur][((g * 4 + t4) * 16 + nl) * 32 + half * 16 + 2 * j];
#pragma unroll
      for (int t4 = 0; t4 < 4; ++t4)
        o[g * 4 + t4] = __builtin_amdgcn_wmma_f32_16x16x32_bf16(
            false, pf.v, false, bv[t4].v, (short)0, o[g * 4 + t4], false, false);
    }
  }

  // normalize + store bf16 [S, nh*128] layout
#pragma unroll
  for (int t8 = 0; t8 < 8; ++t8)
#pragma unroll
    for (int r = 0; r < 8; ++r) {
      int row = m0 + half * 8 + r;
      float v = o[t8][r] / row_l[r];
      out[(size_t)row * 2048 + h * 128 + t8 * 16 + nl] = f2bf(v);
    }
}

extern "C" void kernel_launch(void* const* d_in, const int* in_sizes, int n_in,
                              void* d_out, int out_size, void* d_ws, size_t ws_size,
                              hipStream_t stream) {
  const float* hs = (const float*)d_in[0];
  const float* wq = (const float*)d_in[1];
  const float* wk = (const float*)d_in[2];
  const float* wv = (const float*)d_in[3];
  const float* wo = (const float*)d_in[4];

  const int S = 4096, H = 2048, KV = 512, NQKV = 3072;

  char* ws = (char*)d_ws;
  bf16_t* Xb    = (bf16_t*)ws; ws += (size_t)S * H * 2;
  bf16_t* Wqkvb = (bf16_t*)ws; ws += (size_t)NQKV * H * 2;
  bf16_t* Wob   = (bf16_t*)ws; ws += (size_t)H * H * 2;
  bf16_t* QKVb  = (bf16_t*)ws; ws += (size_t)S * NQKV * 2;
  bf16_t* AOb   = (bf16_t*)ws; ws += (size_t)S * H * 2;

  int n4 = S * H / 4;
  k_f32_to_bf16<<<(n4 + 255) / 256, 256, 0, stream>>>(hs, Xb, n4);
  int nw4 = NQKV * H / 4;
  k_pack_qkv_w<<<(nw4 + 255) / 256, 256, 0, stream>>>(wq, wk, wv, Wqkvb, H * H, KV * H, KV * H);
  n4 = H * H / 4;
  k_f32_to_bf16<<<(n4 + 255) / 256, 256, 0, stream>>>(wo, Wob, n4);

  dim3 g1(S / 128, NQKV / 128);
  k_gemm_bf16<false><<<g1, 256, 0, stream>>>(Xb, Wqkvb, (void*)QKVb, S, NQKV, H);

  int nrope = S * 20 * 64;
  k_rope<<<(nrope + 255) / 256, 256, 0, stream>>>(QKVb, S);

  k_attn<<<16 * (S / 64), 128, 0, stream>>>(QKVb, AOb, S);

  dim3 g2(S / 128, H / 128);
  k_gemm_bf16<true><<<g2, 256, 0, stream>>>(AOb, Wob, d_out, S, H, H);
}
